// Attention_15315853377738
// MI455X (gfx1250) — matrix operand order repaired and checked by reading, not compiled, not run on hardware
//
#include <hip/hip_runtime.h>
#include <hip/hip_bf16.h>

typedef __attribute__((ext_vector_type(16))) _Float16 v16h;
typedef __attribute__((ext_vector_type(8)))  float    v8f;

#define B_   2
#define N_   2048
#define D_   2048
#define H_   16
#define HKV_ 4
#define HD_  128
#define M_   (B_ * N_)      // 4096 rows for all GEMMs

// ---------------------------------------------------------------------------
// WMMA helper: D = A(16x32 f16) * B(32x16 f16) + C(16x16 f32)
// ---------------------------------------------------------------------------
__device__ __forceinline__ v8f wmma_f16(v16h a, v16h b, v8f c) {
    return __builtin_amdgcn_wmma_f32_16x16x32_f16(
        /*neg_a=*/false, a, /*neg_b=*/false, b,
        /*c_mod=*/(short)0, c, /*reuse_a=*/false, /*reuse_b=*/false);
}

// A-matrix fragment (16x32, f16). p -> 32 contiguous halves for this lane's row.
// Lanes 0-15: halves = K{0..7,16..23}; lanes 16-31: K{8..15,24..31}.
__device__ __forceinline__ v16h fragA(const _Float16* p, int hi) {
    v16h r;
    const int kb = hi * 8;
#pragma unroll
    for (int i = 0; i < 8; ++i) {
        r[i]     = p[kb + i];
        r[i + 8] = p[kb + 16 + i];
    }
    return r;
}

// B-matrix fragment (32x16, f16). p -> 32 contiguous halves for this lane's col.
// Lanes 0-15: halves = K{0..7,16..23}; lanes 16-31: K{8..15,24..31}.
__device__ __forceinline__ v16h fragB(const _Float16* p, int hi) {
    v16h r;
    const int kb = hi * 8;
#pragma unroll
    for (int i = 0; i < 8; ++i) {
        r[i]     = p[kb + i];
        r[i + 8] = p[kb + 16 + i];
    }
    return r;
}

// ---------------------------------------------------------------------------
// fp32 -> fp16 bulk convert (one-time per call for x and the weights)
// ---------------------------------------------------------------------------
__global__ __launch_bounds__(256)
void cvt_f16(const float* __restrict__ s, _Float16* __restrict__ d) {
    const int i = (blockIdx.x * blockDim.x + threadIdx.x) * 8;
#pragma unroll
    for (int j = 0; j < 8; ++j) d[i + j] = (_Float16)s[i + j];
}

// ---------------------------------------------------------------------------
// GEMM: C[M,N] = A[M,K] * W[K,N] (+ bias). f16 inputs, f16 or f32 output.
// 256 threads = 8 waves; block tile 128x128; wave tile 32x64; K-step 64
// (16 WMMAs per barrier pair). Staging is pure f16 byte movement.
// ---------------------------------------------------------------------------
template <bool F16OUT>
__global__ __launch_bounds__(256, 1)
void gemm_f16(const _Float16* __restrict__ A, const _Float16* __restrict__ W,
              const float* __restrict__ bias, _Float16* __restrict__ Ch,
              float* __restrict__ Cf, int M, int N, int K, int addBias) {
    __shared__ _Float16 As[128][72];   // [row][k0..63]  (pad to 72)
    __shared__ _Float16 Bs[128][72];   // [col][k0..63]  (W staged transposed)

    const int t    = threadIdx.x;
    const int lane = t & 31;
    const int wave = t >> 5;
    const int l15  = lane & 15;
    const int hi   = (lane >> 4) & 1;

    const int m0 = blockIdx.y * 128;
    const int n0 = blockIdx.x * 128;
    const int wm = (wave >> 1) * 32;   // 4 waves along M
    const int wn = (wave & 1) * 64;    // 2 waves along N

    v8f acc[2][4] = {};

    const int arow = t >> 1, akb = (t & 1) * 32;  // A: 2 thr/row, 32 halves each
    const int wcol = t & 127, wkh = (t >> 7) * 32; // W: thr = col, 32 k each

    for (int k0 = 0; k0 < K; k0 += 64) {
        // A tile: 128 rows x 64 k, contiguous 64B per thread (b128 loads).
        const _Float16* ap = A + (size_t)(m0 + arow) * K + k0 + akb;
#pragma unroll
        for (int i = 0; i < 32; ++i) As[arow][akb + i] = ap[i];

        // W tile: 64 k x 128 cols; per-k the wave reads 64B coalesced rows,
        // each thread writes its column's k-run contiguously into LDS.
        const _Float16* wp = W + (size_t)(k0 + wkh) * N + n0 + wcol;
#pragma unroll
        for (int i = 0; i < 32; ++i) Bs[wcol][wkh + i] = wp[(size_t)i * N];

        if (k0 + 64 < K) {   // prefetch next tiles -> global_prefetch_b8
            __builtin_prefetch(ap + 64, 0, 1);
            __builtin_prefetch(wp + (size_t)64 * N, 0, 1);
        }
        __syncthreads();

#pragma unroll
        for (int kc = 0; kc < 2; ++kc) {
            v16h af[2], bf[4];
#pragma unroll
            for (int mi = 0; mi < 2; ++mi)
                af[mi] = fragA(&As[wm + mi * 16 + l15][kc * 32], hi);
#pragma unroll
            for (int ni = 0; ni < 4; ++ni)
                bf[ni] = fragB(&Bs[wn + ni * 16 + l15][kc * 32], hi);
#pragma unroll
            for (int mi = 0; mi < 2; ++mi)
#pragma unroll
                for (int ni = 0; ni < 4; ++ni)
                    acc[mi][ni] = wmma_f16(af[mi], bf[ni], acc[mi][ni]);
        }
        __syncthreads();
    }

#pragma unroll
    for (int mi = 0; mi < 2; ++mi) {
#pragma unroll
        for (int ni = 0; ni < 4; ++ni) {
            const int row = m0 + wm + mi * 16 + hi * 8;
            const int col = n0 + wn + ni * 16 + l15;
            if (F16OUT) {
#pragma unroll
                for (int r = 0; r < 8; ++r)
                    Ch[(size_t)(row + r) * N + col] = (_Float16)acc[mi][ni][r];
            } else {
                const float bv = addBias ? bias[col] : 0.0f;
#pragma unroll
                for (int r = 0; r < 8; ++r)
                    Cf[(size_t)(row + r) * N + col] = acc[mi][ni][r] + bv;
            }
        }
    }
}

// ---------------------------------------------------------------------------
// RoPE + layout pack: q[B,N,H*HD] f16 -> qh[B,H,N,HD] f16 (rotary applied)
// ---------------------------------------------------------------------------
__global__ void rope_pack_q(const _Float16* __restrict__ qf,
                            const float* __restrict__ cosb,
                            const float* __restrict__ sinb,
                            _Float16* __restrict__ qh) {
    const int i = blockIdx.x * blockDim.x + threadIdx.x;   // B*N*H*64 threads
    const int d2 = i & 63;
    const int h  = (i >> 6) & (H_ - 1);
    const int n  = (i >> 10) & (N_ - 1);
    const int b  = i >> 21;
    const float c = cosb[n * 64 + d2];
    const float s = sinb[n * 64 + d2];
    const size_t src = ((size_t)(b * N_ + n)) * (H_ * HD_) + h * HD_ + d2;
    const float q1 = (float)qf[src], q2 = (float)qf[src + 64];
    const size_t dst = (((size_t)(b * H_ + h)) * N_ + n) * HD_ + d2;
    qh[dst]      = (_Float16)(q1 * c + q2 * s);
    qh[dst + 64] = (_Float16)(q2 * c - q1 * s);
}

// kv[B,N,2*HKV*HD] f16 -> kh (rotary) + vh (copy), both [B,HKV,N,HD] f16
__global__ void rope_pack_kv(const _Float16* __restrict__ kvf,
                             const float* __restrict__ cosb,
                             const float* __restrict__ sinb,
                             _Float16* __restrict__ kh,
                             _Float16* __restrict__ vh) {
    const int i = blockIdx.x * blockDim.x + threadIdx.x;   // B*N*HKV*64 threads
    const int d2 = i & 63;
    const int h  = (i >> 6) & (HKV_ - 1);
    const int n  = (i >> 8) & (N_ - 1);
    const int b  = i >> 19;
    const float c = cosb[n * 64 + d2];
    const float s = sinb[n * 64 + d2];
    const size_t src = ((size_t)(b * N_ + n)) * (2 * HKV_ * HD_) + h * HD_ + d2;
    const float k1 = (float)kvf[src], k2 = (float)kvf[src + 64];
    const size_t dst = (((size_t)(b * HKV_ + h)) * N_ + n) * HD_ + d2;
    kh[dst]      = (_Float16)(k1 * c + k2 * s);
    kh[dst + 64] = (_Float16)(k2 * c - k1 * s);
    vh[dst]      = kvf[src + HKV_ * HD_];
    vh[dst + 64] = kvf[src + HKV_ * HD_ + 64];
}

// ---------------------------------------------------------------------------
// Flash attention: grid (N/64, H, B), block 128 = 4 waves, 16 q-rows per wave.
// KV streamed through LDS in 32-key steps; online softmax; WMMA for S and PV.
// Output written as f16 to aout[B, N, H*HD] feeding the Wo GEMM directly.
// ---------------------------------------------------------------------------
__global__ __launch_bounds__(128, 1)
void attn_fwd(const _Float16* __restrict__ qh, const _Float16* __restrict__ kh,
              const _Float16* __restrict__ vh, const float* __restrict__ mask,
              _Float16* __restrict__ aout) {
    __shared__ _Float16 Kt[32][136];      // [key][d]
    __shared__ _Float16 Vt[128][40];      // [d][key]   (transposed for B frags)
    __shared__ _Float16 Pt[4][16][40];    // per-wave P tile [row][key]

    const int b    = blockIdx.z;
    const int h    = blockIdx.y;
    const int hkv  = h >> 2;              // G = H/HKV = 4
    const int q0   = blockIdx.x * 64;
    const int t    = threadIdx.x;
    const int lane = t & 31;
    const int wave = t >> 5;
    const int l15  = lane & 15;
    const int hi   = (lane >> 4) & 1;

    // Q fragments (A-matrix layout), kept in registers for the whole pass.
    v16h qfrag[4];
    {
        const int myrow = q0 + wave * 16 + l15;
        const _Float16* qp = qh + (((size_t)(b * H_ + h)) * N_ + myrow) * HD_;
#pragma unroll
        for (int kc = 0; kc < 4; ++kc) qfrag[kc] = fragA(qp + kc * 32, hi);
    }

    v8f   oacc[8] = {};
    float mrow[8], lrow[8];
#pragma unroll
    for (int r = 0; r < 8; ++r) { mrow[r] = -3.0e38f; lrow[r] = 0.0f; }

    const _Float16* kbase = kh + ((size_t)(b * HKV_ + hkv)) * N_ * HD_;
    const _Float16* vbase = vh + ((size_t)(b * HKV_ + hkv)) * N_ * HD_;
    const float scale = 0.08838834764831845f;   // HD^-0.5

    for (int j0 = 0; j0 < N_; j0 += 32) {
        // Cooperative stage of 32 keys of K and V (V transposed).
        const int key = t >> 2;
        const int db  = (t & 3) * 32;
        const _Float16* kp = kbase + (size_t)(j0 + key) * HD_ + db;
        const _Float16* vp = vbase + (size_t)(j0 + key) * HD_ + db;
#pragma unroll
        for (int i = 0; i < 32; ++i) Kt[key][db + i] = kp[i];
#pragma unroll
        for (int i = 0; i < 32; ++i) Vt[db + i][key] = vp[i];
        if (j0 + 32 < N_) {
            __builtin_prefetch(kp + 32 * HD_, 0, 1);
            __builtin_prefetch(vp + 32 * HD_, 0, 1);
        }
        __syncthreads();

        // S = Q * K^T for 16 rows x 32 keys (two 16x16 accumulators).
        v8f s0 = {}, s1 = {};
#pragma unroll
        for (int kc = 0; kc < 4; ++kc) {
            v16h kb0 = fragB(&Kt[l15][kc * 32], hi);
            v16h kb1 = fragB(&Kt[16 + l15][kc * 32], hi);
            s0 = wmma_f16(qfrag[kc], kb0, s0);
            s1 = wmma_f16(qfrag[kc], kb1, s1);
        }

        const float bias0 = (1.0f - mask[b * N_ + j0 + l15]) * -1.0e9f;
        const float bias1 = (1.0f - mask[b * N_ + j0 + 16 + l15]) * -1.0e9f;

        // Online softmax; each VGPR index r owns row (r + hi*8), cols on lanes.
#pragma unroll
        for (int r = 0; r < 8; ++r) {
            float a = s0[r] * scale + bias0;
            float c = s1[r] * scale + bias1;
            float mx = fmaxf(a, c);
#pragma unroll
            for (int off = 1; off < 16; off <<= 1)
                mx = fmaxf(mx, __shfl_xor(mx, off, 32));
            const float mnew  = fmaxf(mrow[r], mx);
            const float alpha = __expf(mrow[r] - mnew);
            const float p0 = __expf(a - mnew);
            const float p1 = __expf(c - mnew);
            float rs = p0 + p1;
#pragma unroll
            for (int off = 1; off < 16; off <<= 1)
                rs += __shfl_xor(rs, off, 32);
            lrow[r] = lrow[r] * alpha + rs;
            mrow[r] = mnew;
#pragma unroll
            for (int dt = 0; dt < 8; ++dt) oacc[dt][r] *= alpha;
            const int prow = r + hi * 8;
            Pt[wave][prow][l15]      = (_Float16)p0;
            Pt[wave][prow][16 + l15] = (_Float16)p1;
        }
        __syncthreads();

        // O += P * V  (P re-read through LDS in A-matrix layout).
        v16h pfrag = fragA(&Pt[wave][l15][0], hi);
#pragma unroll
        for (int dt = 0; dt < 8; ++dt) {
            v16h vf = fragB(&Vt[dt * 16 + l15][0], hi);
            oacc[dt] = wmma_f16(pfrag, vf, oacc[dt]);
        }
        __syncthreads();
    }

    // Normalize and write out[B, N, H*HD] as f16.
#pragma unroll
    for (int r = 0; r < 8; ++r) {
        const float inv = 1.0f / lrow[r];
        const int row = q0 + wave * 16 + hi * 8 + r;
        _Float16* op = aout + ((size_t)(b * N_ + row)) * (H_ * HD_) + h * HD_;
#pragma unroll
        for (int dt = 0; dt < 8; ++dt)
            op[dt * 16 + l15] = (_Float16)(oacc[dt][r] * inv);
    }
}

// ---------------------------------------------------------------------------
// Launch: convert -> qkv GEMMs -> rope/pack -> flash attention -> out GEMM
// ---------------------------------------------------------------------------
extern "C" void kernel_launch(void* const* d_in, const int* in_sizes, int n_in,
                              void* d_out, int out_size, void* d_ws, size_t ws_size,
                              hipStream_t stream) {
    const float* x    = (const float*)d_in[0];
    const float* cosb = (const float*)d_in[1];
    const float* sinb = (const float*)d_in[2];
    const float* mask = (const float*)d_in[3];
    const float* Wq   = (const float*)d_in[4];
    const float* Wkv  = (const float*)d_in[5];
    const float* Wo   = (const float*)d_in[6];
    const float* bo   = (const float*)d_in[7];
    float*       out  = (float*)d_out;

    // Workspace carve (~84 MB), all-f16 intermediates. qf16 reused as aoh.
    char* ws = (char*)d_ws;
    _Float16* xh    = (_Float16*)(ws);                       // 16 MB
    _Float16* Wqh   = (_Float16*)(ws + 16777216);            //  8 MB
    _Float16* Wkvh  = (_Float16*)(ws + 25165824);            //  4 MB
    _Float16* Woh   = (_Float16*)(ws + 29360128);            //  8 MB
    _Float16* qf16  = (_Float16*)(ws + 37748736);            // 16 MB (-> aoh)
    _Float16* kvf16 = (_Float16*)(ws + 54525952);            //  8 MB
    _Float16* qhh   = (_Float16*)(ws + 62914560);            // 16 MB
    _Float16* khh   = (_Float16*)(ws + 79691776);            //  4 MB
    _Float16* vhh   = (_Float16*)(ws + 83886080);            //  4 MB
    _Float16* aoh   = qf16;

    // 0) One-time f16 conversions (byte-movement GEMM staging thereafter).
    cvt_f16<<<(M_ * D_) / 2048, 256, 0, stream>>>(x, xh);
    cvt_f16<<<(D_ * H_ * HD_) / 2048, 256, 0, stream>>>(Wq, Wqh);
    cvt_f16<<<(D_ * 2 * HKV_ * HD_) / 2048, 256, 0, stream>>>(Wkv, Wkvh);
    cvt_f16<<<(H_ * HD_ * D_) / 2048, 256, 0, stream>>>(Wo, Woh);

    // 1) Q = x @ Wq   [4096,2048]x[2048,2048] -> f16
    gemm_f16<true><<<dim3(D_ / 128, M_ / 128), 256, 0, stream>>>(
        xh, Wqh, nullptr, qf16, nullptr, M_, D_, D_, 0);
    // 2) KV = x @ Wkv [4096,2048]x[2048,1024] -> f16
    gemm_f16<true><<<dim3((2 * HKV_ * HD_) / 128, M_ / 128), 256, 0, stream>>>(
        xh, Wkvh, nullptr, kvf16, nullptr, M_, 2 * HKV_ * HD_, D_, 0);

    // 3) RoPE + pack to head-major f16
    rope_pack_q<<<(B_ * N_ * H_ * 64) / 256, 256, 0, stream>>>(qf16, cosb, sinb, qhh);
    rope_pack_kv<<<(B_ * N_ * HKV_ * 64) / 256, 256, 0, stream>>>(kvf16, cosb, sinb, khh, vhh);

    // 4) Flash attention -> aoh [B,N,H*HD] f16
    attn_fwd<<<dim3(N_ / 64, H_, B_), 128, 0, stream>>>(qhh, khh, vhh, mask, aoh);

    // 5) out = attnOut @ Wo + bo  (f32 output)
    gemm_f16<false><<<dim3(D_ / 128, M_ / 128), 256, 0, stream>>>(
        aoh, Woh, bo, nullptr, out, M_, D_, H_ * HD_, 1);
}
